// RelFTAttention2D_16097537425938
// MI455X (gfx1250) — matrix operand order repaired and checked by reading, not compiled, not run on hardware
//
#include <hip/hip_runtime.h>
#include <cstdint>
#include <cstddef>

#define CC 128
#define HH 128
#define WW 128
#define BB 8
#define HWSZ (HH*WW)
#define EPSBN 1e-5f
#define SMOOTH 1e-5f

typedef __attribute__((ext_vector_type(16))) _Float16 v16h;
typedef __attribute__((ext_vector_type(8)))  _Float16 v8h;
typedef __attribute__((ext_vector_type(8)))  float    v8f;
typedef __attribute__((ext_vector_type(4)))  float    v4f;

union V16 { v16h v; v8h h[2]; };

// ---------------- conv tile parameters ----------------
#define TWP 32             // pixels (along W) per workgroup
#define XCOLS (TWP+2)      // staged columns incl. halo = 34
#define XP 136             // padded ci stride (halfs) for input staging (16B-aligned rows)
#define WPD 136            // padded ci stride (halfs) for weight staging
#define XS_BYTES  (3*XCOLS*XP*2)            // 27744
#define WSB_BYTES (CC*WPD*2)                // 34816 per buffer
#define SMEM_BYTES (XS_BYTES + 2*WSB_BYTES) // 97376 dynamic LDS

// -------- weight convert: f32 [co][ci][3][3] -> f16 [conv][tap][co][ci] --------
__global__ __launch_bounds__(256) void wconv_kernel(const float* __restrict__ wq,
                                                    const float* __restrict__ wk,
                                                    const float* __restrict__ wv,
                                                    _Float16* __restrict__ wh) {
    int i = blockIdx.x * 256 + threadIdx.x;           // flat over 3*9*128*128
    if (i >= 3 * 9 * CC * CC) return;
    int ci  = i & (CC - 1);
    int co  = (i >> 7) & (CC - 1);
    int q14 = i >> 14;                                 // conv*9 + tap
    int tap = q14 % 9;
    int cv  = q14 / 9;
    const float* src = (cv == 0) ? wq : ((cv == 1) ? wk : wv);
    wh[i] = (_Float16)src[(co * CC + ci) * 9 + tap];
}

// -------- async DMA of one 32KB weight tap tile into an LDS buffer --------
// 2048 16-byte chunks; 256 threads x 8 chunks each; ASYNCcnt-tracked, no VGPR data.
__device__ __forceinline__ void stage_w_async(const _Float16* __restrict__ w9tap,
                                              _Float16* wsbuf, int t) {
    uint32_t ldsbase = (uint32_t)(uintptr_t)wsbuf;     // low 32 bits = LDS byte offset
    #pragma unroll
    for (int j = 0; j < 8; ++j) {
        int idx = j * 256 + t;                         // chunk id 0..2047
        int co  = idx >> 3;                            // row (co)
        int cf  = (idx & 7) << 3;                      // half offset within row
        uint32_t lds  = ldsbase + (uint32_t)(co * WPD + cf) * 2u;
        uint32_t goff = (uint32_t)(co * CC + cf) * 2u; // byte offset from w9tap
        asm volatile("global_load_async_to_lds_b128 %0, %1, %2"
                     :: "v"(lds), "v"(goff), "s"(w9tap) : "memory");
    }
}

// -------- conv3x3 + BN + sigmoid via implicit GEMM on v_wmma_f32_16x16x32_f16 --------
// Weight tiles double-buffered in LDS; tap t+1 DMA'd (async-to-LDS) while tap t computes.
__global__ __launch_bounds__(256) void conv_bn_sig_kernel(
    const float* __restrict__ x,
    const _Float16* __restrict__ w9,       // [9][co=128][ci=128] f16
    const float* __restrict__ gamma, const float* __restrict__ beta,
    const float* __restrict__ mean,  const float* __restrict__ var,
    float* __restrict__ out)
{
    extern __shared__ __align__(16) char smem[];
    _Float16* Xs    = (_Float16*)smem;                  // [3][34][XP]
    _Float16* WsBuf = (_Float16*)(smem + XS_BYTES);     // 2 x [128][WPD]

    const int bx = blockIdx.x;
    const int tw = bx & 3;                    // W tile (4 tiles of 32)
    const int h  = (bx >> 2) & (HH - 1);
    const int b  = bx >> 9;
    const int w0 = tw * TWP;
    const int t  = threadIdx.x;
    const int lane = t & 31;
    const int wv   = t >> 5;                  // wave id 0..7
    const int lm   = lane & 15;
    const int hlf  = lane >> 4;

    // kick off DMA of tap-0 weights; it overlaps the register-path input staging
    stage_w_async(w9, WsBuf, t);

    // ---- stage input patch (f32 -> f16, transposed to [row][col][ci]) ----
    for (int e = t; e < 3 * XCOLS * CC; e += 256) {
        int cc = e % XCOLS;
        int rc = e / XCOLS;
        int ci = rc & (CC - 1);
        int r  = rc >> 7;
        int hy = h + r - 1;
        int wx = w0 - 1 + cc;
        float vin = 0.f;
        if (hy >= 0 && hy < HH && wx >= 0 && wx < WW)
            vin = x[(((size_t)b * CC + ci) * HH + hy) * WW + wx];
        Xs[(r * XCOLS + cc) * XP + ci] = (_Float16)vin;
    }

    asm volatile("s_wait_asynccnt 0x0" ::: "memory");
    __syncthreads();

    const int mt = wv & 1;                    // M tile (16 pixels each)
    const int ng = wv >> 1;                   // N group of 32 channels (0..3)

    v8f acc[2] = {};                          // two 16x16 f32 accumulators

    for (int tap = 0; tap < 9; ++tap) {
        const _Float16* Wcur = WsBuf + (size_t)(tap & 1) * (CC * WPD);
        if (tap < 8)  // prefetch next tap into the other buffer while we compute
            stage_w_async(w9 + (size_t)(tap + 1) * CC * CC,
                          WsBuf + (size_t)((tap + 1) & 1) * (CC * WPD), t);

        const int kh = tap / 3, kw = tap - 3 * kh;
        const _Float16* aRow = &Xs[(kh * XCOLS + (mt * 16 + lm + kw)) * XP];

        #pragma unroll
        for (int c0 = 0; c0 < CC; c0 += 32) {
            // A 16x32: lane l -> M=l%16; K = 8*(l/16)+0..7 then 16+8*(l/16)+0..7
            V16 A;
            A.h[0] = *(const v8h*)(aRow + c0 + 8 * hlf);
            A.h[1] = *(const v8h*)(aRow + c0 + 16 + 8 * hlf);
            #pragma unroll
            for (int j = 0; j < 2; ++j) {
                // B 32x16: lane l -> N=l%16; K = 8*(l/16)+0..7 then 16+8*(l/16)+0..7
                const _Float16* bp = &Wcur[(ng * 32 + j * 16 + lm) * WPD + c0 + 8 * hlf];
                V16 Bm;
                Bm.h[0] = *(const v8h*)bp;
                Bm.h[1] = *(const v8h*)(bp + 16);
                acc[j] = __builtin_amdgcn_wmma_f32_16x16x32_f16(
                    false, A.v, false, Bm.v, (short)0, acc[j], false, false);
            }
        }

        asm volatile("s_wait_asynccnt 0x0" ::: "memory");
        __syncthreads();
    }

    // ---- epilogue: BN + sigmoid, store f32 ----
    const int wbase = w0 + mt * 16 + 8 * hlf;   // M = r + 8*(lane/16) per C/D layout
    #pragma unroll
    for (int j = 0; j < 2; ++j) {
        int co = ng * 32 + j * 16 + lm;
        float sc = gamma[co] * rsqrtf(var[co] + EPSBN);
        float sh = beta[co] - mean[co] * sc;
        float* o = out + (((size_t)b * CC + co) * HH + h) * WW + wbase;
        v4f lo, hi;
        #pragma unroll
        for (int r = 0; r < 4; ++r) {
            lo[r] = 1.f / (1.f + __expf(-(acc[j][r]     * sc + sh)));
            hi[r] = 1.f / (1.f + __expf(-(acc[j][r + 4] * sc + sh)));
        }
        *(v4f*)(o)     = lo;
        *(v4f*)(o + 4) = hi;
    }
}

// -------- fractal Tanimoto helpers --------
__device__ inline float tnmt_base(float tpl, float tpp, float tll) {
    float num = tpl + SMOOTH;
    float den = 0.f;
    float a = 1.f;
    #pragma unroll
    for (int d = 0; d < 5; ++d) {
        float bb = -(2.f * a - 1.f);
        den += 1.f / (a * (tpp + tll) + bb * tpl + SMOOTH);
        a *= 2.f;
    }
    return num * den * 0.2f;
}
__device__ inline float ftanimoto(float tpl, float tpp, float tll,
                                  float sp, float sl, float N) {
    float t0 = tnmt_base(tpl, tpp, tll);
    float tplc = N - sp - sl + tpl;
    float tppc = N - 2.f * sp + tpp;
    float tllc = N - 2.f * sl + tll;
    return 0.5f * (t0 + tnmt_base(tplc, tppc, tllc));
}

// -------- channel attention: reduce over H*W per (b,c) --------
__global__ __launch_bounds__(256) void att_chan_kernel(const float* __restrict__ q,
                                                       const float* __restrict__ k,
                                                       float* __restrict__ att) {
    int bc = blockIdx.x;                       // b*CC + c
    const v4f* qp = (const v4f*)(q + (size_t)bc * HWSZ);
    const v4f* kp = (const v4f*)(k + (size_t)bc * HWSZ);
    float tpl = 0, tpp = 0, tll = 0, sp = 0, sl = 0;
    for (int i = threadIdx.x; i < HWSZ / 4; i += 256) {
        v4f qv = qp[i], kv = kp[i];
        #pragma unroll
        for (int r = 0; r < 4; ++r) {
            float a = qv[r], bq = kv[r];
            tpl += a * bq; tpp += a * a; tll += bq * bq; sp += a; sl += bq;
        }
    }
    for (int off = 16; off; off >>= 1) {
        tpl += __shfl_down(tpl, off); tpp += __shfl_down(tpp, off);
        tll += __shfl_down(tll, off); sp  += __shfl_down(sp, off);
        sl  += __shfl_down(sl, off);
    }
    __shared__ float red[5][8];
    int lane = threadIdx.x & 31, wv = threadIdx.x >> 5;
    if (lane == 0) { red[0][wv]=tpl; red[1][wv]=tpp; red[2][wv]=tll; red[3][wv]=sp; red[4][wv]=sl; }
    __syncthreads();
    if (threadIdx.x == 0) {
        float s[5];
        #pragma unroll
        for (int j = 0; j < 5; ++j) { float a = 0; for (int i = 0; i < 8; ++i) a += red[j][i]; s[j] = a; }
        att[bc] = ftanimoto(s[0], s[1], s[2], s[3], s[4], (float)HWSZ);
    }
}

// -------- spatial attention: reduce over C per (b,h,w) --------
__global__ __launch_bounds__(256) void att_spat_kernel(const float* __restrict__ q,
                                                       const float* __restrict__ k,
                                                       float* __restrict__ att) {
    int p  = blockIdx.x * 256 + threadIdx.x;   // b*HW + hw
    int b  = p >> 14;
    int hw = p & (HWSZ - 1);
    const float* qp = q + (size_t)b * CC * HWSZ + hw;
    const float* kp = k + (size_t)b * CC * HWSZ + hw;
    float tpl = 0, tpp = 0, tll = 0, sp = 0, sl = 0;
    #pragma unroll 4
    for (int c = 0; c < CC; ++c) {
        float a = qp[(size_t)c * HWSZ], bq = kp[(size_t)c * HWSZ];
        tpl += a * bq; tpp += a * a; tll += bq * bq; sp += a; sl += bq;
    }
    att[p] = ftanimoto(tpl, tpp, tll, sp, sl, (float)CC);
}

// -------- final: 0.5*(att_chan + att_spat)*v then BN row 3 --------
__global__ __launch_bounds__(256) void final_kernel(const float* __restrict__ v,
                                                    const float* __restrict__ attC,
                                                    const float* __restrict__ attS,
                                                    const float* __restrict__ gamma,
                                                    const float* __restrict__ beta,
                                                    const float* __restrict__ mean,
                                                    const float* __restrict__ var,
                                                    float* __restrict__ out) {
    size_t e = ((size_t)blockIdx.x * 256 + threadIdx.x) * 4;
    int hw = (int)(e & (HWSZ - 1));
    int c  = (int)((e >> 14) & (CC - 1));
    int b  = (int)(e >> 21);
    float sc = gamma[c] * rsqrtf(var[c] + EPSBN);
    float sh = beta[c] - mean[c] * sc;
    float ac = attC[b * CC + c];
    v4f vv = *(const v4f*)(v + e);
    v4f as = *(const v4f*)(attS + ((size_t)b << 14) + hw);
    v4f o;
    #pragma unroll
    for (int r = 0; r < 4; ++r)
        o[r] = (0.5f * (ac + as[r]) * vv[r]) * sc + sh;
    *(v4f*)(out + e) = o;
}

// ---------------- workspace layout (bytes) ----------------
#define WS_Q    ((size_t)0)
#define WS_K    ((size_t)67108864)
#define WS_V    ((size_t)134217728)
#define WS_WH   ((size_t)201326592)
#define WS_AC   (WS_WH + (size_t)884736)
#define WS_AS   (WS_AC + (size_t)4096)

extern "C" void kernel_launch(void* const* d_in, const int* in_sizes, int n_in,
                              void* d_out, int out_size, void* d_ws, size_t ws_size,
                              hipStream_t stream) {
    const float* in1 = (const float*)d_in[0];
    const float* in2 = (const float*)d_in[1];
    const float* in3 = (const float*)d_in[2];
    const float* wq  = (const float*)d_in[3];
    const float* wk  = (const float*)d_in[4];
    const float* wvv = (const float*)d_in[5];
    const float* g   = (const float*)d_in[6];   // [4][C]
    const float* be  = (const float*)d_in[7];
    const float* mn  = (const float*)d_in[8];
    const float* vr  = (const float*)d_in[9];

    char* ws = (char*)d_ws;
    float*    Q    = (float*)(ws + WS_Q);
    float*    K    = (float*)(ws + WS_K);
    float*    V    = (float*)(ws + WS_V);
    _Float16* WH   = (_Float16*)(ws + WS_WH);
    float*    ATTC = (float*)(ws + WS_AC);
    float*    ATTS = (float*)(ws + WS_AS);

    // allow >64KB dynamic LDS for the conv kernel (gfx1250: 320KB/WGP)
    (void)hipFuncSetAttribute((const void*)conv_bn_sig_kernel,
                              hipFuncAttributeMaxDynamicSharedMemorySize, SMEM_BYTES);

    wconv_kernel<<<(3 * 9 * CC * CC + 255) / 256, 256, 0, stream>>>(wq, wk, wvv, WH);

    const int convGrid = BB * HH * (WW / TWP);  // 4096
    conv_bn_sig_kernel<<<convGrid, 256, SMEM_BYTES, stream>>>(in1, WH,                g,          be,          mn,          vr,          Q);
    conv_bn_sig_kernel<<<convGrid, 256, SMEM_BYTES, stream>>>(in2, WH + 9 * CC * CC,  g + CC,     be + CC,     mn + CC,     vr + CC,     K);
    conv_bn_sig_kernel<<<convGrid, 256, SMEM_BYTES, stream>>>(in3, WH + 18 * CC * CC, g + 2 * CC, be + 2 * CC, mn + 2 * CC, vr + 2 * CC, V);

    att_chan_kernel<<<BB * CC, 256, 0, stream>>>(Q, K, ATTC);
    att_spat_kernel<<<(BB * HWSZ) / 256, 256, 0, stream>>>(Q, K, ATTS);

    final_kernel<<<(BB * CC * HWSZ) / 1024, 256, 0, stream>>>(
        V, ATTC, ATTS, g + 3 * CC, be + 3 * CC, mn + 3 * CC, vr + 3 * CC, (float*)d_out);
}
